// ModulatedLatentAttention_53480932770313
// MI455X (gfx1250) — compile-verified, matrix operand order checked by reading, not run on hardware
//
#include <hip/hip_runtime.h>
#include <hip/hip_bf16.h>
#include <math.h>

// Shapes (fixed by the reference)
#define BB 8
#define UU 512
#define VV 2048
#define DD 512
#define CC 384
#define HH 8
#define HD 64
#define NR (BB*UU)   // 4096 receiver rows
#define NS (BB*VV)   // 16384 sender rows

typedef __attribute__((ext_vector_type(16))) __bf16 v16bf;
typedef __attribute__((ext_vector_type(8)))  float  v8f;

union FragA { v16bf v; unsigned int u32[8]; };

__device__ __forceinline__ unsigned short f2bf(float f) {
  unsigned int x = __float_as_uint(f);
  unsigned int r = (x + 0x7FFFu + ((x >> 16) & 1u)) >> 16;  // RNE
  return (unsigned short)r;
}
__device__ __forceinline__ float bf2f(unsigned short u) {
  return __uint_as_float(((unsigned int)u) << 16);
}
// k-base for element-pair j of a 16-bit A/B fragment (ISA 7.12.2 layout)
__device__ __forceinline__ int frag_kbase(int j, int half) {
  return ((j < 4) ? (2 * j) : (16 + 2 * (j - 4))) + half * 8;
}
// low 32 bits of a flat pointer into LDS == LDS byte offset (aperture layout)
__device__ __forceinline__ unsigned lds_addr32(const void* p) {
  return (unsigned)(unsigned long long)p;
}
// CDNA5 async global->LDS copy, 16 bytes per lane (GVS addressing), ASYNCcnt
__device__ __forceinline__ void async_b128(unsigned lds, unsigned voff_bytes,
                                           const void* sbase) {
  asm volatile("global_load_async_to_lds_b128 %0, %1, %2"
               :: "v"(lds), "v"(voff_bytes), "s"(sbase) : "memory");
}
__device__ __forceinline__ void wait_async0() {
  asm volatile("s_wait_asynccnt 0" ::: "memory");
}

// ---------------- cast f32 -> bf16 ----------------
__global__ void mla_cast_kernel(const float* __restrict__ src,
                                unsigned short* __restrict__ dst, int n) {
  int i = blockIdx.x * blockDim.x + threadIdx.x;
  if (i < n) dst[i] = f2bf(src[i]);
}
// cast + transpose weights: src [K,512] f32 -> dst [512,K] bf16
__global__ void mla_castT_kernel(const float* __restrict__ src,
                                 unsigned short* __restrict__ dst, int K) {
  int idx = blockIdx.x * blockDim.x + threadIdx.x;
  if (idx < K * DD) {
    int k = idx >> 9, n = idx & 511;
    dst[(long)n * K + k] = f2bf(src[idx]);
  }
}

// ---------------- fused LayerNorm -> bf16 ----------------
__global__ __launch_bounds__(256)
void mla_ln_kernel(const float* __restrict__ x, const float* __restrict__ w,
                   const float* __restrict__ bb, unsigned short* __restrict__ y) {
  __shared__ float s1[256], s2[256];
  long row = blockIdx.x;
  int tid = threadIdx.x;
  float v0 = x[row * DD + tid];
  float v1 = x[row * DD + 256 + tid];
  s1[tid] = v0 + v1;
  s2[tid] = v0 * v0 + v1 * v1;
  __syncthreads();
  for (int s = 128; s > 0; s >>= 1) {
    if (tid < s) { s1[tid] += s1[tid + s]; s2[tid] += s2[tid + s]; }
    __syncthreads();
  }
  float mu   = s1[0] * (1.0f / DD);
  float var  = s2[0] * (1.0f / DD) - mu * mu;
  float rstd = rsqrtf(var + 1e-5f);
  y[row * DD + tid]       = f2bf((v0 - mu) * rstd * w[tid]       + bb[tid]);
  y[row * DD + 256 + tid] = f2bf((v1 - mu) * rstd * w[tid + 256] + bb[tid + 256]);
}

// ---------------- WMMA GEMM, N fixed = 512, async double-buffered ----------------
// A: [M,K] bf16 row-major.  Bt: [512,K] bf16 (pre-transposed weights).
// MODE 0: out_bf16 = mult * (1 + acc + bias)      (FiLM modulation)
// MODE 1: out_bf16 = acc + bias                   (Q/K/V projection)
// MODE 2: out_f32  = resid + gamma*(acc + bias)   (output proj + layerscale + residual)
#define ALS 40   // LDS row stride (ushorts): 80B, 16B-aligned chunks, conflict-skewed
template <int MODE>
__global__ __launch_bounds__(256)
void mla_gemm_kernel(const unsigned short* __restrict__ A,
                     const unsigned short* __restrict__ Bt,
                     int K,
                     const float* __restrict__ bias,          // [512]
                     const unsigned short* __restrict__ mult, // [M,512] (MODE 0)
                     const float* __restrict__ resid,         // [M,512] (MODE 2)
                     const float* __restrict__ gamma,         // [512]   (MODE 2)
                     unsigned short* __restrict__ outb,
                     float* __restrict__ outf) {
  __shared__ unsigned short Als[2][128 * ALS];
  __shared__ unsigned short Bls[2][64 * ALS];
  const int tid  = threadIdx.x;
  const int lane = tid & 31, wave = tid >> 5;
  const int half = lane >> 4, lr = lane & 15;
  const long bm = (long)blockIdx.x * 128;
  const int  bn = blockIdx.y * 64;

  v8f acc[4];
  for (int j = 0; j < 4; ++j)
    for (int r = 0; r < 8; ++r) acc[j][r] = 0.0f;

  // issue one 128x32 A tile + 64x32 B tile as async B128 copies
  auto issueTile = [&](int bufi, int k0) {
    unsigned short* Ab = &Als[bufi][0];
    unsigned short* Bb = &Bls[bufi][0];
#pragma unroll
    for (int t = 0; t < 2; ++t) {          // A: 512 x 16B chunks
      int i = tid + t * 256;
      int r = i >> 2, c4 = i & 3;
      async_b128(lds_addr32(&Ab[r * ALS + c4 * 8]),
                 (unsigned)(((bm + r) * (long)K + k0 + c4 * 8) * 2), A);
    }
    {                                      // B: 256 x 16B chunks
      int c = tid >> 2, c4 = tid & 3;
      async_b128(lds_addr32(&Bb[c * ALS + c4 * 8]),
                 (unsigned)((((long)(bn + c)) * K + k0 + c4 * 8) * 2), Bt);
    }
  };

  const int nk = K / 32;
  issueTile(0, 0);
  for (int t = 0; t < nk; ++t) {
    wait_async0();
    __syncthreads();
    if (t + 1 < nk) issueTile((t + 1) & 1, (t + 1) * 32);
    const unsigned short* Ab = &Als[t & 1][0];
    const unsigned short* Bb = &Bls[t & 1][0];

    FragA af;
    {
      int m = wave * 16 + lr;
#pragma unroll
      for (int j = 0; j < 8; ++j)
        af.u32[j] = *(const unsigned int*)&Ab[m * ALS + frag_kbase(j, half)];
    }
#pragma unroll
    for (int jt = 0; jt < 4; ++jt) {
      FragA bf;
      int n = jt * 16 + lr;
#pragma unroll
      for (int j = 0; j < 8; ++j)
        bf.u32[j] = *(const unsigned int*)&Bb[n * ALS + frag_kbase(j, half)];
      acc[jt] = __builtin_amdgcn_wmma_f32_16x16x32_bf16(
          false, af.v, false, bf.v, (short)0, acc[jt], false, false);
    }
  }

#pragma unroll
  for (int jt = 0; jt < 4; ++jt) {
    int col = bn + jt * 16 + lr;
#pragma unroll
    for (int r = 0; r < 8; ++r) {
      long row = bm + wave * 16 + r + half * 8;
      long idx = row * DD + col;
      float v = acc[jt][r];
      if (MODE == 0) {
        outb[idx] = f2bf(bf2f(mult[idx]) * (1.0f + v + bias[col]));
      } else if (MODE == 1) {
        outb[idx] = f2bf(v + bias[col]);
      } else {
        outf[idx] = resid[idx] + gamma[col] * (v + bias[col]);
      }
    }
  }
}

// ---------------- flash attention (per b, h, 64-row U tile) ----------------
#define KVS 80   // K/V LDS row stride (ushorts): 160B, 16B-aligned chunks
__global__ __launch_bounds__(128)
void mla_attn_kernel(const unsigned short* __restrict__ Q,
                     const unsigned short* __restrict__ Kb,
                     const unsigned short* __restrict__ Vb,
                     unsigned short* __restrict__ Msg) {
  __shared__ unsigned short Kls[2][64 * KVS];   // [key][hd]
  __shared__ unsigned short Vls[2][64 * KVS];   // [key][hd] (row-major)
  __shared__ unsigned short Pls[4 * 16 * 64];   // per-wave P staging
  const int tid = threadIdx.x, lane = tid & 31, wave = tid >> 5;
  const int half = lane >> 4, lr = lane & 15;
  const int ub = blockIdx.x, h = blockIdx.y, b = blockIdx.z;
  const long qrow0 = (long)b * UU + ub * 64 + wave * 16;
  const int hc = h * HD;

  // Q fragments resident for the whole kernel (2 k-steps of 32 over HD=64)
  FragA aq[2];
  {
    long m = qrow0 + lr;
#pragma unroll
    for (int ks = 0; ks < 2; ++ks)
#pragma unroll
      for (int j = 0; j < 8; ++j)
        aq[ks].u32[j] =
            *(const unsigned int*)&Q[m * DD + hc + ks * 32 + frag_kbase(j, half)];
  }

  auto issueChunk = [&](int bufi, int c) {
    long key0 = (long)b * VV + c * 64;
#pragma unroll
    for (int t = 0; t < 4; ++t) {    // 512 x 16B chunks each for K and V
      int i = tid + t * 128;
      int r = i >> 3, c4 = i & 7;
      unsigned voff = (unsigned)(((key0 + r) * DD + hc + c4 * 8) * 2);
      async_b128(lds_addr32(&Kls[bufi][r * KVS + c4 * 8]), voff, Kb);
      async_b128(lds_addr32(&Vls[bufi][r * KVS + c4 * 8]), voff, Vb);
    }
  };

  float mrow[8], lrow[8];
  for (int r = 0; r < 8; ++r) { mrow[r] = -1e30f; lrow[r] = 0.0f; }
  v8f o[4];
  for (int j = 0; j < 4; ++j)
    for (int r = 0; r < 8; ++r) o[j][r] = 0.0f;

  issueChunk(0, 0);
  for (int c = 0; c < VV / 64; ++c) {
    wait_async0();
    __syncthreads();
    if (c + 1 < VV / 64) issueChunk((c + 1) & 1, c + 1);
    const unsigned short* Kt = &Kls[c & 1][0];
    const unsigned short* Vt = &Vls[c & 1][0];

    // S = Q @ K^T (scaled)
    v8f s[4];
    for (int j = 0; j < 4; ++j)
      for (int r = 0; r < 8; ++r) s[j][r] = 0.0f;
#pragma unroll
    for (int ks = 0; ks < 2; ++ks) {
#pragma unroll
      for (int jt = 0; jt < 4; ++jt) {
        FragA bk;
        int n = jt * 16 + lr;
#pragma unroll
        for (int j = 0; j < 8; ++j)
          bk.u32[j] =
              *(const unsigned int*)&Kt[n * KVS + ks * 32 + frag_kbase(j, half)];
        s[jt] = __builtin_amdgcn_wmma_f32_16x16x32_bf16(
            false, aq[ks].v, false, bk.v, (short)0, s[jt], false, false);
      }
    }
    for (int j = 0; j < 4; ++j)
      for (int r = 0; r < 8; ++r) s[j][r] *= 0.125f;  // 1/sqrt(64)

    // online softmax; row r (+8*half) lives across the 16 lanes of each half
    float corr[8];
#pragma unroll
    for (int r = 0; r < 8; ++r) {
      float xm = fmaxf(fmaxf(s[0][r], s[1][r]), fmaxf(s[2][r], s[3][r]));
      for (int msk = 1; msk < 16; msk <<= 1)
        xm = fmaxf(xm, __shfl_xor(xm, msk, 32));
      float mnew = fmaxf(mrow[r], xm);
      float c0 = __expf(mrow[r] - mnew);
      float ps = 0.0f;
      for (int j = 0; j < 4; ++j) {
        float p = __expf(s[j][r] - mnew);
        s[j][r] = p;
        ps += p;
      }
      for (int msk = 1; msk < 16; msk <<= 1) ps += __shfl_xor(ps, msk, 32);
      lrow[r] = lrow[r] * c0 + ps;
      mrow[r] = mnew;
      corr[r] = c0;
    }
    for (int j = 0; j < 4; ++j)
      for (int r = 0; r < 8; ++r) o[j][r] *= corr[r];

    // stage P (C-layout) -> LDS -> A-layout fragments (wave-local)
    unsigned short* Pw = &Pls[wave * 1024];
    for (int j = 0; j < 4; ++j)
      for (int r = 0; r < 8; ++r)
        Pw[(r + half * 8) * 64 + j * 16 + lr] = f2bf(s[j][r]);

    // O += P @ V   (V row-major in LDS -> gather fragment pairs as u16)
#pragma unroll
    for (int ks = 0; ks < 2; ++ks) {
      FragA ap;
#pragma unroll
      for (int j = 0; j < 8; ++j)
        ap.u32[j] =
            *(const unsigned int*)&Pw[lr * 64 + ks * 32 + frag_kbase(j, half)];
#pragma unroll
      for (int jt = 0; jt < 4; ++jt) {
        FragA bv;
        int n = jt * 16 + lr;
#pragma unroll
        for (int j = 0; j < 8; ++j) {
          int k0 = ks * 32 + frag_kbase(j, half);
          unsigned lo = Vt[(k0 + 0) * KVS + n];
          unsigned hi = Vt[(k0 + 1) * KVS + n];
          bv.u32[j] = lo | (hi << 16);
        }
        o[jt] = __builtin_amdgcn_wmma_f32_16x16x32_bf16(
            false, ap.v, false, bv.v, (short)0, o[jt], false, false);
      }
    }
  }

  // finalize: divide by row sums, write into [B*U, 512] msg layout
  for (int r = 0; r < 8; ++r) {
    float inv = 1.0f / lrow[r];
    long row = qrow0 + r + half * 8;
    for (int jt = 0; jt < 4; ++jt) {
      int col = hc + jt * 16 + lr;
      Msg[row * DD + col] = f2bf(o[jt][r] * inv);
    }
  }
}

// ---------------- host-side orchestration ----------------
static unsigned short* ws_take(char*& p, size_t elems) {
  unsigned short* r = (unsigned short*)p;
  p += ((elems * sizeof(unsigned short)) + 255) & ~(size_t)255;
  return r;
}

extern "C" void kernel_launch(void* const* d_in, const int* in_sizes, int n_in,
                              void* d_out, int out_size, void* d_ws, size_t ws_size,
                              hipStream_t stream) {
  const float* r_states = (const float*)d_in[0];
  const float* r_codes  = (const float*)d_in[1];
  const float* s_states = (const float*)d_in[2];
  const float* s_codes  = (const float*)d_in[3];
  const float* ln_r_w = (const float*)d_in[4];
  const float* ln_r_b = (const float*)d_in[5];
  const float* ln_s_w = (const float*)d_in[6];
  const float* ln_s_b = (const float*)d_in[7];
  const float* q_w  = (const float*)d_in[8];
  const float* q_b  = (const float*)d_in[9];
  const float* q_mw = (const float*)d_in[10];
  const float* q_mb = (const float*)d_in[11];
  const float* k_w  = (const float*)d_in[12];
  const float* k_b  = (const float*)d_in[13];
  const float* k_mw = (const float*)d_in[14];
  const float* k_mb = (const float*)d_in[15];
  const float* v_w  = (const float*)d_in[16];
  const float* v_b  = (const float*)d_in[17];
  const float* v_mw = (const float*)d_in[18];
  const float* v_mb = (const float*)d_in[19];
  const float* e_w  = (const float*)d_in[20];
  const float* e_b  = (const float*)d_in[21];
  const float* e_mw = (const float*)d_in[22];
  const float* e_mb = (const float*)d_in[23];
  const float* gamma = (const float*)d_in[24];
  float* out = (float*)d_out;

  char* p = (char*)d_ws;
  unsigned short* rc  = ws_take(p, (size_t)NR * CC);
  unsigned short* sc  = ws_take(p, (size_t)NS * CC);
  unsigned short* rln = ws_take(p, (size_t)NR * DD);
  unsigned short* sln = ws_take(p, (size_t)NS * DD);
  unsigned short* wq  = ws_take(p, (size_t)DD * DD);   // transposed [512][512]
  unsigned short* wk  = ws_take(p, (size_t)DD * DD);
  unsigned short* wv  = ws_take(p, (size_t)DD * DD);
  unsigned short* we  = ws_take(p, (size_t)DD * DD);
  unsigned short* mwq = ws_take(p, (size_t)CC * DD);   // transposed [512][384]
  unsigned short* mwk = ws_take(p, (size_t)CC * DD);
  unsigned short* mwv = ws_take(p, (size_t)CC * DD);
  unsigned short* mwe = ws_take(p, (size_t)CC * DD);
  unsigned short* xbuf = ws_take(p, (size_t)NS * DD);  // reused for xq/xk/xv/xm
  unsigned short* qb  = ws_take(p, (size_t)NR * DD);
  unsigned short* kb  = ws_take(p, (size_t)NS * DD);
  unsigned short* vb  = ws_take(p, (size_t)NS * DD);
  unsigned short* msg = ws_take(p, (size_t)NR * DD);

  auto cast = [&](const float* src, unsigned short* dst, int n) {
    mla_cast_kernel<<<(n + 255) / 256, 256, 0, stream>>>(src, dst, n);
  };
  auto castT = [&](const float* src, unsigned short* dst, int K) {
    mla_castT_kernel<<<(K * DD + 255) / 256, 256, 0, stream>>>(src, dst, K);
  };
  cast(r_codes, rc, NR * CC);
  cast(s_codes, sc, NS * CC);
  castT(q_w, wq, DD);   castT(k_w, wk, DD);
  castT(v_w, wv, DD);   castT(e_w, we, DD);
  castT(q_mw, mwq, CC); castT(k_mw, mwk, CC);
  castT(v_mw, mwv, CC); castT(e_mw, mwe, CC);

  mla_ln_kernel<<<NR, 256, 0, stream>>>(r_states, ln_r_w, ln_r_b, rln);
  mla_ln_kernel<<<NS, 256, 0, stream>>>(s_states, ln_s_w, ln_s_b, sln);

  dim3 gR(NR / 128, DD / 64), gS(NS / 128, DD / 64);
  // Q path
  mla_gemm_kernel<0><<<gR, 256, 0, stream>>>(rc, mwq, CC, q_mb, rln, nullptr, nullptr, xbuf, nullptr);
  mla_gemm_kernel<1><<<gR, 256, 0, stream>>>(xbuf, wq, DD, q_b, nullptr, nullptr, nullptr, qb, nullptr);
  // K path
  mla_gemm_kernel<0><<<gS, 256, 0, stream>>>(sc, mwk, CC, k_mb, sln, nullptr, nullptr, xbuf, nullptr);
  mla_gemm_kernel<1><<<gS, 256, 0, stream>>>(xbuf, wk, DD, k_b, nullptr, nullptr, nullptr, kb, nullptr);
  // V path
  mla_gemm_kernel<0><<<gS, 256, 0, stream>>>(sc, mwv, CC, v_mb, sln, nullptr, nullptr, xbuf, nullptr);
  mla_gemm_kernel<1><<<gS, 256, 0, stream>>>(xbuf, wv, DD, v_b, nullptr, nullptr, nullptr, vb, nullptr);

  // flash attention: grid (U/64, H, B)
  mla_attn_kernel<<<dim3(UU / 64, HH, BB), 128, 0, stream>>>(qb, kb, vb, msg);

  // output projection: FiLM-modulate msg, then GEMM with layerscale + residual
  mla_gemm_kernel<0><<<gR, 256, 0, stream>>>(rc, mwe, CC, e_mb, msg, nullptr, nullptr, xbuf, nullptr);
  mla_gemm_kernel<2><<<gR, 256, 0, stream>>>(xbuf, we, DD, e_b, nullptr, r_states, gamma, nullptr, out);
}
